// NeRFCamera_51049981281458
// MI455X (gfx1250) — compile-verified, matrix operand order checked by reading, not run on hardware
//
#include <hip/hip_runtime.h>

typedef __attribute__((ext_vector_type(16))) _Float16 v16h;
typedef __attribute__((ext_vector_type(8)))  float    v8f;

#define NB     2
#define HW     65536              // 256*256
#define NRAYS  (NB*HW)            // 131072
#define PTS    64
#define IMP    64
#define TPB    64                 // 2 waves/block; LDS ~34.5KB
#define FOCAL_INV (1.0f/4.2f)
#define NEARZ  0.1f
#define FARZ   2.0f
#define DSTEP  ((FARZ-NEARZ)/63.0f)
#define F_OFS  (NB*3*HW)          // 393216
#define D_OFS  (2*NB*3*HW)        // 786432

__device__ __forceinline__ float sigm(float x){ return 1.0f/(1.0f+__expf(-x)); }

// camera-plane direction for a ray id (z component is 1.0)
__device__ __forceinline__ void camdir(int ray, float& cx, float& cy){
  int p = ray & (HW-1);
  int i = p >> 8, j = p & 255;
  const float s = 2.0f/255.0f;                 // linspace(1,-1,256)
  cx = (1.0f - (float)j*s)*FOCAL_INV;
  cy = (1.0f - (float)i*s)*FOCAL_INV;
}

__global__ void init_mm(unsigned* mm){
  mm[0] = 0x7F800000u;   // +inf bits  (running min)
  mm[1] = 0u;            // 0.0f bits  (running max)  -- depths are strictly positive
}

__global__ __launch_bounds__(TPB) void nerf_main(const float* __restrict__ tm,
                                                 const float* __restrict__ Wq,
                                                 float* __restrict__ out,
                                                 unsigned* __restrict__ mm)
{
  __shared__ float s_w [PTS][TPB];   // coarse weights (per-thread column)
  __shared__ float s_fd[IMP][TPB];   // fine depths    (per-thread column)
  __shared__ float s_bg[TPB][8];     // WMMA result: [g0..3 | base'0..3] per ray
  __shared__ float s_red[2][TPB];    // min/max reduction

  const int tid  = threadIdx.x;
  const int ray  = blockIdx.x*TPB + tid;
  const int bb   = ray >> 16;                    // batch (uniform per wave: 65536 % 32 == 0)
  const int pix  = ray & (HW-1);
  const int lane = tid & 31;                     // wave32
  const int wr0  = blockIdx.x*TPB + (tid & ~31); // first ray of this wave

  // ---------------- per-wave WMMA setup: [g|base'] = cam @ (R^T W) for 32 rays ----------------
  const int col = lane & 15;                     // output channel (0..7 live)
  // M[j][col] = sum_i R[i][j] * w_i ; w_i = Wq[i][col] (g, col<4) or Wq[3+i][col-4] (base')
  float M0 = 0.0f, M1 = 0.0f, M2 = 0.0f;
  if (col < 8) {
    const int wc = col & 3;
    const int wr = (col < 4) ? 0 : 3;
    #pragma unroll
    for (int i = 0; i < 3; ++i) {
      float wi = Wq[(wr+i)*4 + wc];
      M0 += tm[bb*12 + i*4 + 0]*wi;
      M1 += tm[bb*12 + i*4 + 1]*wi;
      M2 += tm[bb*12 + i*4 + 2]*wi;
    }
  }
  v16h bfr, afr0, afr1;
  #pragma unroll
  for (int i = 0; i < 16; ++i) { bfr[i] = (_Float16)0.0f; afr0[i] = (_Float16)0.0f; afr1[i] = (_Float16)0.0f; }
  float cx0, cy0, cx1, cy1;
  camdir(wr0 + col,      cx0, cy0);              // tile 0: rays wr0+0..15   (row = col here)
  camdir(wr0 + 16 + col, cx1, cy1);              // tile 1: rays wr0+16..31
  if (lane < 16) {                               // lanes 0..15 hold K=0..7 halves; K=0..2 live
    bfr[0]  = (_Float16)M0;  bfr[1]  = (_Float16)M1;  bfr[2]  = (_Float16)M2;
    afr0[0] = (_Float16)cx0; afr0[1] = (_Float16)cy0; afr0[2] = (_Float16)1.0f;
    afr1[0] = (_Float16)cx1; afr1[1] = (_Float16)cy1; afr1[2] = (_Float16)1.0f;
  }
  v8f cz;
  #pragma unroll
  for (int i = 0; i < 8; ++i) cz[i] = 0.0f;
  v8f dt0 = __builtin_amdgcn_wmma_f32_16x16x32_f16(false, afr0, false, bfr, (short)0, cz, false, false);
  v8f dt1 = __builtin_amdgcn_wmma_f32_16x16x32_f16(false, afr1, false, bfr, (short)0, cz, false, false);

  // scatter D (row = ray-in-tile, col = channel) to per-ray LDS
  {
    const int wl   = tid & ~31;
    const int mofs = 8*(lane >> 4);
    #pragma unroll
    for (int v = 0; v < 8; ++v) {
      if (col < 8) {
        s_bg[wl +      (mofs + v)][col] = dt0[v];
        s_bg[wl + 16 + (mofs + v)][col] = dt1[v];
      }
    }
  }
  __syncthreads();

  float g0 = s_bg[tid][0], g1 = s_bg[tid][1], g2 = s_bg[tid][2], g3 = s_bg[tid][3];
  float base0 = s_bg[tid][4], base1 = s_bg[tid][5], base2 = s_bg[tid][6], base3 = s_bg[tid][7];
  // add origin contribution: base += t_b . W[0:3]
  #pragma unroll
  for (int j = 0; j < 3; ++j) {
    float tj = tm[bb*12 + j*4 + 3];
    base0 += tj*Wq[j*4+0]; base1 += tj*Wq[j*4+1];
    base2 += tj*Wq[j*4+2]; base3 += tj*Wq[j*4+3];
  }

  // ---------------- coarse pass: transmittance scan, store w ----------------
  float T = 1.0f, a0 = 0.0f, a1 = 0.0f, a2 = 0.0f, sw = 0.0f;
  for (int p = 0; p < PTS; ++p) {
    float d  = NEARZ + (float)p*DSTEP;
    float f0 = base0 + d*g0, f1 = base1 + d*g1, f2 = base2 + d*g2, f3 = base3 + d*g3;
    float op = sigm(f0);
    float w  = op*T;
    T *= (1.0f - op);
    a0 += w*sigm(f1); a1 += w*sigm(f2); a2 += w*sigm(f3);
    sw += w;
    s_w[p][tid] = w;
  }
  out[bb*3*HW + 0*HW + pix] = a0;
  out[bb*3*HW + 1*HW + pix] = a1;
  out[bb*3*HW + 2*HW + pix] = a2;

  // ---------------- CDF inverse sampling -> 64 fine depths ----------------
  {
    const float invS = 1.0f/(sw + (float)PTS*1e-5f);
    int   ind     = 0;
    float cdf_lo  = 0.0f;
    float cdf_hi  = (s_w[0][tid] + 1e-5f)*invS;  // cdfs[ind]
    float binPrev = 0.0f;
    for (int k = 0; k <= IMP; ++k) {
      float u = (float)k*(1.0f/(float)IMP);
      while (ind < PTS && cdf_hi <= u) {
        ++ind;
        cdf_lo = cdf_hi;
        if (ind < PTS) cdf_hi += (s_w[ind][tid] + 1e-5f)*invS;
      }
      int below, above; float c0, c1;
      if (ind == 0)        { below = 0;      above = 0;      c0 = cdf_hi; c1 = cdf_hi; }
      else if (ind >= PTS) { below = PTS-1;  above = PTS-1;  c0 = cdf_lo; c1 = cdf_lo; }
      else                 { below = ind-1;  above = ind;    c0 = cdf_lo; c1 = cdf_hi; }
      float dd0 = NEARZ + (float)below*DSTEP;
      float dd1 = NEARZ + (float)above*DSTEP;
      float den = c1 - c0; if (den < 1e-8f) den = 1.0f;
      float t   = (u - c0)/den;
      t = fminf(fmaxf(t, 0.0f), 1.0f);
      float bin = dd0 + t*(dd1 - dd0);
      if (k > 0) s_fd[k-1][tid] = 0.5f*(binPrev + bin);
      binPrev = bin;
    }
  }

  // ---------------- merge (stable) + fine scan ----------------
  float T2 = 1.0f, b0 = 0.0f, b1 = 0.0f, b2 = 0.0f, sw2 = 0.0f, dsum = 0.0f;
  {
    int ic = 0, iq = 0;
    for (int m = 0; m < PTS + IMP; ++m) {
      float dc = (ic < PTS) ? (NEARZ + (float)ic*DSTEP) : 3.0e38f;
      float dq = (iq < IMP) ? s_fd[iq][tid]             : 3.0e38f;
      float d;
      if (dc <= dq) { d = dc; ++ic; } else { d = dq; ++iq; }   // coarse-first on ties (stable sort)
      float f0 = base0 + d*g0, f1 = base1 + d*g1, f2 = base2 + d*g2, f3 = base3 + d*g3;
      float op = sigm(f0);
      float w  = op*T2;
      T2 *= (1.0f - op);
      b0 += w*sigm(f1); b1 += w*sigm(f2); b2 += w*sigm(f3);
      sw2 += w; dsum += w*d;
    }
  }
  float aop    = fminf(fmaxf(sw2, 0.0f), 1.0f);
  float rdepth = dsum + (1.0f - aop)*FARZ;      // d_all.max() == FAR == 2.0 globally

  out[F_OFS + bb*3*HW + 0*HW + pix] = b0;
  out[F_OFS + bb*3*HW + 1*HW + pix] = b1;
  out[F_OFS + bb*3*HW + 2*HW + pix] = b2;
  out[D_OFS + ray] = rdepth;                    // un-normalized; pass 2 rescales in place

  // ---------------- block min/max -> global atomics (positive float == monotone uint) ----------
  s_red[0][tid] = rdepth;
  s_red[1][tid] = rdepth;
  __syncthreads();
  for (int s = TPB/2; s > 0; s >>= 1) {
    if (tid < s) {
      s_red[0][tid] = fminf(s_red[0][tid], s_red[0][tid+s]);
      s_red[1][tid] = fmaxf(s_red[1][tid], s_red[1][tid+s]);
    }
    __syncthreads();
  }
  if (tid == 0) {
    atomicMin(&mm[0], __float_as_uint(s_red[0][0]));
    atomicMax(&mm[1], __float_as_uint(s_red[1][0]));
  }
}

__global__ void norm_depth(float* __restrict__ out, const unsigned* __restrict__ mm)
{
  int i = blockIdx.x*blockDim.x + threadIdx.x;
  if (i >= NRAYS) return;
  float mn = __uint_as_float(mm[0]);
  float mx = __uint_as_float(mm[1]);
  float v  = out[D_OFS + i];
  out[D_OFS + i] = (v - mn)/(mx - mn);
}

extern "C" void kernel_launch(void* const* d_in, const int* in_sizes, int n_in,
                              void* d_out, int out_size, void* d_ws, size_t ws_size,
                              hipStream_t stream)
{
  const float* tm = (const float*)d_in[0];   // (2,3,4) transform matrices
  const float* Wq = (const float*)d_in[1];   // (6,4) query weights
  float*    out   = (float*)d_out;           // c_img | f_img | d_img  (917504 f32)
  unsigned* mm    = (unsigned*)d_ws;         // 2 uints: min/max bits

  init_mm   <<<dim3(1),           dim3(1),   0, stream>>>(mm);
  nerf_main <<<dim3(NRAYS/TPB),   dim3(TPB), 0, stream>>>(tm, Wq, out, mm);
  norm_depth<<<dim3(NRAYS/256),   dim3(256), 0, stream>>>(out, mm);
}
